// PairConLossWithNeighbors_42752104465182
// MI455X (gfx1250) — matrix operand order repaired and checked by reading, not compiled, not run on hardware
//
#include <hip/hip_runtime.h>

// ---------------------------------------------------------------------------
// PairConLossWithNeighbors on MI455X (gfx1250, wave32, WMMA)
//
// sim = fn @ fn^T with f32 -> bf16(hi)+bf16(lo) split:
//   sim ~= Ah*Bh + Ah*Bl + Al*Bh     (3x v_wmma_f32_16x16x32_bf16 per K=32)
// Fused row reduction: per-row sum of exp(sim/T) over off-diagonal + per-row
// top-5 off-diagonal exp values (diag is always the row max, so the reference
// zeroes exactly {diag, top-5 offdiag}).  ng = sum - top5sum.
// B tiles are double-buffered in LDS and staged with CDNA5
// GLOBAL_LOAD_ASYNC_TO_LDS_B128 (ASYNCcnt) overlapping the WMMA compute.
// Tile loop unrolled by 2 so the ping-pong buffer index is compile-time
// constant -> pure ds_load_b128 base+imm addressing in the hot loop.
// ---------------------------------------------------------------------------

typedef __bf16 bf16;
typedef __bf16 v16bf __attribute__((ext_vector_type(16)));
typedef __bf16 v8bf  __attribute__((ext_vector_type(8)));
typedef __bf16 v4bf  __attribute__((ext_vector_type(4)));
typedef float  v8f   __attribute__((ext_vector_type(8)));

#define NROWS   8192
#define BHALF   4096
#define D       512
#define TINV    20.0f          // 1/TEMPERATURE
#define WAVES   2              // waves per workgroup (64 threads)
#define NCHUNK  4              // column splits (each WG sweeps 2048 cols)
#define TILES_PER_CHUNK (NROWS / 16 / NCHUNK)   // 128 (even)
#define STR     520            // padded LDS row stride (bf16 elems)

#if defined(__has_builtin)
#if __has_builtin(__builtin_amdgcn_global_load_async_to_lds_b128)
#define HAVE_ASYNC_LDS 1
#endif
#endif

// builtin parameter type (from hipcc diagnostic): int __vector(4) *
typedef int gv4i __attribute__((vector_size(16)));
typedef __attribute__((address_space(1))) gv4i gv4i_glob;
typedef __attribute__((address_space(3))) gv4i gv4i_lds;

__device__ __forceinline__ void cp16_to_lds(const bf16* gsrc, bf16* ldst) {
#ifdef HAVE_ASYNC_LDS
    __builtin_amdgcn_global_load_async_to_lds_b128(
        (gv4i_glob*)(gsrc), (gv4i_lds*)(ldst), 0, 0);
#else
    *(uint4*)ldst = *(const uint4*)gsrc;
#endif
}

__device__ __forceinline__ void wait_async0() {
#ifdef HAVE_ASYNC_LDS
#if __has_builtin(__builtin_amdgcn_s_wait_asynccnt)
    __builtin_amdgcn_s_wait_asynccnt(0);
#else
    asm volatile("s_wait_asynccnt 0x0" ::: "memory");
#endif
#endif
}

// ---------------- workspace layout (bytes) ----------------
// fnh : [8192][512] bf16          8 MiB    @ 0
// fnl : [8192][512] bf16          8 MiB    @ 8Mi
// ngp : [4][8192]   f32         128 KiB    @ 16Mi
// t5p : [4][8192][5] f32        640 KiB    @ 16Mi + 128Ki
// pos : [4096]      f32          16 KiB    @ 16Mi + 768Ki

__global__ __launch_bounds__(128) void prep_kernel(
    const float* __restrict__ f1, const float* __restrict__ f2,
    bf16* __restrict__ fnh, bf16* __restrict__ fnl, float* __restrict__ pos)
{
    const int row = blockIdx.x;
    const int tid = threadIdx.x;                 // 128 threads x 4 elems = 512
    const float* src = (row < BHALF) ? (f1 + (size_t)row * D)
                                     : (f2 + (size_t)(row - BHALF) * D);
    float4 x = ((const float4*)src)[tid];
    float ss = x.x*x.x + x.y*x.y + x.z*x.z + x.w*x.w;
    float dp = 0.0f;
    if (row < BHALF) {
        float4 y = ((const float4*)(f2 + (size_t)row * D))[tid];
        dp = x.x*y.x + x.y*y.y + x.z*y.z + x.w*y.w;
    }
    #pragma unroll
    for (int off = 16; off > 0; off >>= 1) {
        ss += __shfl_xor(ss, off, 32);
        dp += __shfl_xor(dp, off, 32);
    }
    __shared__ float red[8];
    if ((tid & 31) == 0) { red[tid >> 5] = ss; red[4 + (tid >> 5)] = dp; }
    __syncthreads();
    ss = red[0] + red[1] + red[2] + red[3];
    dp = red[4] + red[5] + red[6] + red[7];
    const float scale = 1.0f / fmaxf(sqrtf(ss), 1e-8f);

    float e[4] = { x.x, x.y, x.z, x.w };
    v4bf hv, lv;
    #pragma unroll
    for (int q = 0; q < 4; ++q) {
        float xn = e[q] * scale;
        bf16 h = (bf16)xn;
        hv[q] = h;
        lv[q] = (bf16)(xn - (float)h);
    }
    *(v4bf*)(fnh + (size_t)row * D + tid * 4) = hv;
    *(v4bf*)(fnl + (size_t)row * D + tid * 4) = lv;

    if (row < BHALF && tid == 0) pos[row] = __expf(dp * TINV);
}

__global__ __launch_bounds__(WAVES * 32) void simreduce_kernel(
    const bf16* __restrict__ fnh, const bf16* __restrict__ fnl,
    float* __restrict__ ngp, float* __restrict__ t5p)
{
    __shared__ __align__(16) bf16 Ah[WAVES][16][STR];
    __shared__ __align__(16) bf16 Al[WAVES][16][STR];
    __shared__ __align__(16) bf16 Bh[2][16][STR];   // double-buffered
    __shared__ __align__(16) bf16 Bl[2][16][STR];

    const int tid  = threadIdx.x;
    const int wv   = tid >> 5;
    const int lane = tid & 31;
    const int m    = lane & 15;      // A-row / B-col within tile
    const int hw   = lane >> 4;      // half-wave

    const int rb        = blockIdx.x * (WAVES * 16) + wv * 16;
    const int myRowTile = rb >> 4;
    const int ct0       = blockIdx.y * TILES_PER_CHUNK;

    float s[8];                      // per-row exp-sum partials
    float t[8][5];                   // per-row top-5 exp partials (descending)
    #pragma unroll
    for (int j = 0; j < 8; ++j) {
        s[j] = 0.0f;
        #pragma unroll
        for (int q = 0; q < 5; ++q) t[j][q] = 0.0f;
    }

    // stage B tile 'cb' (hi+lo) into buffer 'buf', all 64 threads, 16B chunks
    auto loadB = [&](int buf, int cb) {
        for (int c = tid; c < 16 * D / 8; c += WAVES * 32) {
            int r = c >> 6;
            int k = (c & 63) << 3;
            cp16_to_lds(&fnh[(size_t)(cb + r) * D + k], &Bh[buf][r][k]);
            cp16_to_lds(&fnl[(size_t)(cb + r) * D + k], &Bl[buf][r][k]);
        }
    };

    // one tile: prefetch tile ti+1 into buffer cur^1, compute from buffer cur.
    // 'cur' is always a literal at the call site so LDS addressing is static.
    auto tile_body = [&](int ti, int cur) {
        const int ct = ct0 + ti;
        if (ti + 1 < TILES_PER_CHUNK) loadB(cur ^ 1, (ct + 1) << 4);

        v8f acc = {};
        #pragma unroll 4
        for (int k0 = 0; k0 < D; k0 += 32) {
            union { v16bf v; v8bf h[2]; } ah, al, bh, bl;
            // A 16x32 bf16 layout: elems[0..7]=K(k0+8*hw..), elems[8..15]=K(k0+16+8*hw..)
            ah.h[0] = *(const v8bf*)&Ah[wv][m][k0 + 8 * hw];
            ah.h[1] = *(const v8bf*)&Ah[wv][m][k0 + 16 + 8 * hw];
            al.h[0] = *(const v8bf*)&Al[wv][m][k0 + 8 * hw];
            al.h[1] = *(const v8bf*)&Al[wv][m][k0 + 16 + 8 * hw];
            // B 32x16 bf16 layout: elems[0..7]=K(k0+8*hw..), elems[8..15]=K(k0+16+8*hw..) at col m
            bh.h[0] = *(const v8bf*)&Bh[cur][m][k0 + 8 * hw];
            bh.h[1] = *(const v8bf*)&Bh[cur][m][k0 + 16 + 8 * hw];
            bl.h[0] = *(const v8bf*)&Bl[cur][m][k0 + 8 * hw];
            bl.h[1] = *(const v8bf*)&Bl[cur][m][k0 + 16 + 8 * hw];
            acc = __builtin_amdgcn_wmma_f32_16x16x32_bf16(false, ah.v, false, bh.v, (short)0, acc, false, false);
            acc = __builtin_amdgcn_wmma_f32_16x16x32_bf16(false, ah.v, false, bl.v, (short)0, acc, false, false);
            acc = __builtin_amdgcn_wmma_f32_16x16x32_bf16(false, al.v, false, bh.v, (short)0, acc, false, false);
        }

        const bool diagTile = (ct == myRowTile);
        #pragma unroll
        for (int j = 0; j < 8; ++j) {           // C layout: VGPR j -> M = j + 8*hw, N = m
            float v = acc[j];
            float e = __expf(TINV * v);
            bool isd = diagTile && (m == (j + (hw << 3)));
            e = isd ? 0.0f : e;
            s[j] += e;
            float x = e;                        // branchless sorted-5 insert
            #pragma unroll
            for (int q = 0; q < 5; ++q) {
                float mx = fmaxf(t[j][q], x);
                x = fminf(t[j][q], x);
                t[j][q] = mx;
            }
        }

        wait_async0();        // our prefetch for ti+1 has landed in LDS
        __syncthreads();      // publish to all waves; orders reads of old buf
    };

    // stage this wave's 16-row A panel (hi+lo) into LDS, 16B chunks
    for (int c = lane; c < 16 * D / 8; c += 32) {
        int r = c >> 6;
        int k = (c & 63) << 3;
        *(uint4*)&Ah[wv][r][k] = *(const uint4*)&fnh[(size_t)(rb + r) * D + k];
        *(uint4*)&Al[wv][r][k] = *(const uint4*)&fnl[(size_t)(rb + r) * D + k];
    }

    // prologue: fill buffer 0 with the first tile
    loadB(0, ct0 << 4);
    wait_async0();
    __syncthreads();

    for (int tp = 0; tp < TILES_PER_CHUNK; tp += 2) {
        tile_body(tp,     0);   // literal buffer index -> static LDS offsets
        tile_body(tp + 1, 1);
    }

    // merge across the 16 lanes of each half-wave (halves stay disjoint under xor<16)
    #pragma unroll
    for (int off = 1; off < 16; off <<= 1) {
        #pragma unroll
        for (int j = 0; j < 8; ++j) {
            s[j] += __shfl_xor(s[j], off, 32);
            float in[5];
            #pragma unroll
            for (int q = 0; q < 5; ++q) in[q] = __shfl_xor(t[j][q], off, 32);
            #pragma unroll
            for (int q = 0; q < 5; ++q) {
                float x = in[q];
                #pragma unroll
                for (int p = 0; p < 5; ++p) {
                    float mx = fmaxf(t[j][p], x);
                    x = fminf(t[j][p], x);
                    t[j][p] = mx;
                }
            }
        }
    }

    if (m == 0) {                               // lanes 0 and 16 each own 8 rows
        #pragma unroll
        for (int j = 0; j < 8; ++j) {
            int row = rb + (hw << 3) + j;
            size_t base = (size_t)blockIdx.y * NROWS + row;
            ngp[base] = s[j];
            #pragma unroll
            for (int q = 0; q < 5; ++q) t5p[base * 5 + q] = t[j][q];
        }
    }
}

__global__ __launch_bounds__(256) void finalize_kernel(
    const float* __restrict__ ngp, const float* __restrict__ t5p,
    const float* __restrict__ pos, float* __restrict__ out)
{
    const int tid = threadIdx.x;
    float acc = 0.0f;
    for (int row = tid; row < NROWS; row += 256) {
        float total = 0.0f;
        float t5[5] = {0.f, 0.f, 0.f, 0.f, 0.f};
        #pragma unroll
        for (int c = 0; c < NCHUNK; ++c) {
            size_t base = (size_t)c * NROWS + row;
            total += ngp[base];
            #pragma unroll
            for (int q = 0; q < 5; ++q) {
                float x = t5p[base * 5 + q];
                #pragma unroll
                for (int p = 0; p < 5; ++p) {
                    float mx = fmaxf(t5[p], x);
                    x = fminf(t5[p], x);
                    t5[p] = mx;
                }
            }
        }
        float ng = total - (t5[0] + t5[1] + t5[2] + t5[3] + t5[4]);
        float p  = pos[row & (BHALF - 1)];
        acc += logf((ng + p) / p);              // -log(pos/(ng+pos))
    }
    #pragma unroll
    for (int off = 16; off > 0; off >>= 1) acc += __shfl_xor(acc, off, 32);
    __shared__ float red[8];
    if ((tid & 31) == 0) red[tid >> 5] = acc;
    __syncthreads();
    if (tid == 0) {
        float tt = 0.0f;
        #pragma unroll
        for (int i = 0; i < 8; ++i) tt += red[i];
        out[0] = tt / (float)NROWS;
    }
}

extern "C" void kernel_launch(void* const* d_in, const int* in_sizes, int n_in,
                              void* d_out, int out_size, void* d_ws, size_t ws_size,
                              hipStream_t stream)
{
    (void)in_sizes; (void)n_in; (void)out_size; (void)ws_size;
    const float* f1 = (const float*)d_in[0];
    const float* f2 = (const float*)d_in[1];

    char* w = (char*)d_ws;
    bf16*  fnh = (bf16*)w;
    bf16*  fnl = (bf16*)(w + ((size_t)8 << 20));
    float* ngp = (float*)(w + ((size_t)16 << 20));
    float* t5p = (float*)(w + ((size_t)16 << 20) + ((size_t)128 << 10));
    float* pos = (float*)(w + ((size_t)16 << 20) + ((size_t)768 << 10));

    prep_kernel<<<NROWS, 128, 0, stream>>>(f1, f2, fnh, fnl, pos);

    dim3 grid(NROWS / (WAVES * 16), NCHUNK);
    simreduce_kernel<<<grid, WAVES * 32, 0, stream>>>(fnh, fnl, ngp, t5p);

    finalize_kernel<<<1, 256, 0, stream>>>(ngp, t5p, pos, (float*)d_out);
}
